// pointer_6545530159660
// MI455X (gfx1250) — matrix operand order repaired and checked by reading, not compiled, not run on hardware
//
#include <hip/hip_runtime.h>

typedef __attribute__((ext_vector_type(16))) _Float16 v16h;
typedef __attribute__((ext_vector_type(8)))  _Float16 v8h;
typedef __attribute__((ext_vector_type(4)))  _Float16 v4h;
typedef __attribute__((ext_vector_type(8)))  float    v8f;

#define ASTR 264                         // padded row stride in f16 elements (256 + 8)
// LDS layout (dynamic)
#define SMEM_W_OFF    0                  // 256 x ASTR f16 = 135168 B
#define SMEM_A_OFF    135168             // 32 x ASTR f16  = 16896 B
#define SMEM_BIAS_OFF 152064             // 256 f32
#define SMEM_VW_OFF   153088             // 256 f32
#define SMEM_ROW_OFF  154112             // 128 f32
#define SMEM_BYTES    154624
// workspace layout
#define WS_DEC_OFF    0                  // 32*256 f32 = 32768 B
#define WS_WT_OFF     32768              // 256 x ASTR f16 = 135168 B

__device__ __forceinline__ float fast_tanh(float x) {
    x = fminf(15.f, fmaxf(-15.f, x));
    float e = __expf(2.f * x);
    return (e - 1.f) * __builtin_amdgcn_rcpf(e + 1.f);   // v_exp_f32 + v_rcp_f32
}
__device__ __forceinline__ float sigmoidf(float x) {
    x = fminf(30.f, fmaxf(-30.f, x));
    return __builtin_amdgcn_rcpf(1.f + __expf(-x));
}

// Add src shifted right by <shr> lanes within each 16-lane DPP row (zero-filled).
template <int SHR>
__device__ __forceinline__ float dpp_row_shr_add(float s) {
    int v = __builtin_amdgcn_update_dpp(0, __float_as_int(s),
                                        0x110 | SHR, /*row_mask*/0xF, /*bank_mask*/0xF,
                                        /*bound_ctrl(zero-fill)*/ true);
    return s + __int_as_float(v);
}

// ------------- Kernel 0: W_e -> transposed, padded f16 Wt (one-time) -------------
__global__ __launch_bounds__(256) void wprep_kernel(const float* __restrict__ W_e,
                                                    _Float16* __restrict__ Wt)
{
    const int idx = blockIdx.x * 256 + threadIdx.x;   // over (n,k), k fastest: coalesced writes
    const int n = idx >> 8, k = idx & 255;
    Wt[n * ASTR + k] = (_Float16)W_e[k * 256 + n];
}

// ---------------- Kernel 1: LSTM step + dec_proj (tiny) ----------------
__global__ __launch_bounds__(256) void lstm_dec_kernel(
    const float* __restrict__ h0, const float* __restrict__ c0,
    const float* __restrict__ start_token,
    const float* __restrict__ W_k, const float* __restrict__ W_r,
    const float* __restrict__ b_lstm,
    const float* __restrict__ W_d, const float* __restrict__ b_d,
    float* __restrict__ dec_out)
{
    const int b = blockIdx.x;
    const int j = threadIdx.x;              // 0..255
    __shared__ float xs[256], hs[256], hout[256];
    xs[j] = start_token[b * 256 + j];
    hs[j] = h0[b * 256 + j];
    __syncthreads();

    float zi = b_lstm[j], zf = b_lstm[256 + j], zg = b_lstm[512 + j], zo = b_lstm[768 + j];
    for (int k = 0; k < 256; ++k) {
        const float xk = xs[k], hk = hs[k];
        const float* wk = W_k + k * 1024;
        const float* wr = W_r + k * 1024;
        zi += xk * wk[j]       + hk * wr[j];
        zf += xk * wk[256 + j] + hk * wr[256 + j];
        zg += xk * wk[512 + j] + hk * wr[512 + j];
        zo += xk * wk[768 + j] + hk * wr[768 + j];
    }
    const float c = sigmoidf(zf) * c0[b * 256 + j] + sigmoidf(zi) * fast_tanh(zg);
    const float h = sigmoidf(zo) * fast_tanh(c);
    hout[j] = h;
    __syncthreads();

    float d = b_d[j];
    for (int k = 0; k < 256; ++k)
        d += hout[k] * W_d[k * 256 + j];
    dec_out[b * 256 + j] = d;
}

// ---- Kernel 2: fused enc_proj GEMM (f16 WMMA) + tanh + v_w reduction ----
// grid = (S/128, B), block = 256 (8 waves). Each block: 128 rows (4 chunks of 32) x N=256.
__global__ __launch_bounds__(256) void attn_gemm_kernel(
    const float* __restrict__ enc, const _Float16* __restrict__ Wt,
    const float* __restrict__ b_e, const float* __restrict__ v_w,
    const float* __restrict__ v_b, const float* __restrict__ dec,
    float* __restrict__ scores)
{
    extern __shared__ __align__(16) char smem_raw[];
    _Float16* Wsm   = (_Float16*)(smem_raw + SMEM_W_OFF);    // [256][ASTR] : Wsm[n][k] = W_e[k][n]
    _Float16* Asm   = (_Float16*)(smem_raw + SMEM_A_OFF);    // [32][ASTR] rows = s
    float*   biasS  = (float*)(smem_raw + SMEM_BIAS_OFF);    // b_e[n] + dec_proj[b][n]
    float*   vwS    = (float*)(smem_raw + SMEM_VW_OFF);
    float*   rowsum = (float*)(smem_raw + SMEM_ROW_OFF);     // [128]

    const int tid = threadIdx.x;
    const int b   = blockIdx.y;
    const int s0  = blockIdx.x * 128;
    const float* encBlk = enc + ((size_t)b * 8192 + s0) * 256;

    // Stage Wt (pre-converted f16, pre-padded) into LDS: straight 16B memcpy
    {
        const v8h* wsrc = (const v8h*)Wt;
        v8h* wdst = (v8h*)Wsm;
        #pragma unroll 1
        for (int i = tid; i < (256 * ASTR) / 8; i += 256)
            wdst[i] = wsrc[i];
    }
    biasS[tid] = b_e[tid] + dec[b * 256 + tid];
    vwS[tid]   = v_w[tid];
    if (tid < 128) rowsum[tid] = 0.f;

    const int w    = tid >> 5;
    const int lane = tid & 31;
    const int hi   = lane >> 4;     // half-wave select
    const int lo   = lane & 15;

    #pragma unroll 1
    for (int cc = 0; cc < 4; ++cc) {
        __syncthreads();   // previous chunk's readers of Asm are done
        // Stage A chunk: 32 rows x 256, float4 loads -> packed 4xf16 LDS stores
        const float4* asrc = (const float4*)(encBlk + cc * 32 * 256);
        #pragma unroll
        for (int it = 0; it < 8; ++it) {
            const int i = tid + 256 * it;           // 2048 float4s
            const float4 f = asrc[i];
            const int r = i >> 6;                   // row (i*4)/256
            const int k = (i & 63) * 4;
            v4h h4 = { (_Float16)f.x, (_Float16)f.y, (_Float16)f.z, (_Float16)f.w };
            *(v4h*)(Asm + r * ASTR + k) = h4;
        }
        // Prefetch next chunk's A rows while this chunk computes
        if (cc < 3) {
            const float* nxt = encBlk + (cc + 1) * 32 * 256;
            #pragma unroll
            for (int it = 0; it < 8; ++it)
                __builtin_prefetch(nxt + (tid + 256 * it) * 4, 0, 3);
        }
        __syncthreads();

        v8f acc[2][2] = {};             // [m-tile][j] ; wave's N-tile j -> nt = w + 8*j

        #pragma unroll 2
        for (int kk = 0; kk < 8; ++kk) {
            const int kb = kk * 32;
            v16h afrag[2], bfrag[2];
            // A fragment (16x32 f16): lanes 0-15 K=kb+0..7/kb+16..23, lanes 16-31 +8
            #pragma unroll
            for (int mt = 0; mt < 2; ++mt) {
                const _Float16* ap = Asm + (16 * mt + lo) * ASTR + kb + 8 * hi;
                v8h a0 = *(const v8h*)ap;
                v8h a1 = *(const v8h*)(ap + 16);
                afrag[mt] = __builtin_shufflevector(a0, a1, 0,1,2,3,4,5,6,7,8,9,10,11,12,13,14,15);
            }
            // B fragment (32x16 f16): lane -> column n, K = kb+0..7/kb+16..23, lanes 16-31 +8
            #pragma unroll
            for (int j = 0; j < 2; ++j) {
                const int nt = w + 8 * j;
                const _Float16* bp = Wsm + (16 * nt + lo) * ASTR + kb + 8 * hi;
                v8h b0 = *(const v8h*)bp;
                v8h b1 = *(const v8h*)(bp + 16);
                bfrag[j] = __builtin_shufflevector(b0, b1, 0,1,2,3,4,5,6,7,8,9,10,11,12,13,14,15);
            }
            #pragma unroll
            for (int mt = 0; mt < 2; ++mt)
                #pragma unroll
                for (int j = 0; j < 2; ++j)
                    acc[mt][j] = __builtin_amdgcn_wmma_f32_16x16x32_f16(
                        false, afrag[mt], false, bfrag[j],
                        (short)0, acc[mt][j], false, false);
        }

        // Epilogue: rowsum[row] += sum_n tanh(acc + bias[n]) * v_w[n]
        // C layout: VGPR v -> M = 16*mt + 8*hi + v ; N = 16*nt + lo
        #pragma unroll
        for (int mt = 0; mt < 2; ++mt) {
            #pragma unroll
            for (int v = 0; v < 8; ++v) {
                float s = 0.f;
                #pragma unroll
                for (int j = 0; j < 2; ++j) {
                    const int n = 16 * (w + 8 * j) + lo;
                    const float e = acc[mt][j][v] + biasS[n];
                    s += fast_tanh(e) * vwS[n];
                }
                // VALU-only prefix-sum within each 16-lane DPP row; lane lo==15 holds total
                s = dpp_row_shr_add<1>(s);
                s = dpp_row_shr_add<2>(s);
                s = dpp_row_shr_add<4>(s);
                s = dpp_row_shr_add<8>(s);
                if (lo == 15)
                    atomicAdd(&rowsum[32 * cc + 16 * mt + 8 * hi + v], s);
            }
        }
    }
    __syncthreads();
    if (tid < 128)
        scores[(size_t)b * 8192 + s0 + tid] = rowsum[tid] + v_b[0];
}

// ---------------- Kernel 3: in-place softmax over S per batch ----------------
__global__ __launch_bounds__(256) void softmax_kernel(float* __restrict__ out)
{
    const int b = blockIdx.x;
    const int tid = threadIdx.x;
    float* row = out + (size_t)b * 8192;
    __shared__ float red[256];

    float vals[32];
    float m = -3.4e38f;
    #pragma unroll
    for (int i = 0; i < 32; ++i) {
        vals[i] = row[tid + 256 * i];
        m = fmaxf(m, vals[i]);
    }
    red[tid] = m; __syncthreads();
    for (int s = 128; s > 0; s >>= 1) {
        if (tid < s) red[tid] = fmaxf(red[tid], red[tid + s]);
        __syncthreads();
    }
    m = red[0]; __syncthreads();

    float sum = 0.f;
    #pragma unroll
    for (int i = 0; i < 32; ++i) {
        vals[i] = __expf(vals[i] - m);
        sum += vals[i];
    }
    red[tid] = sum; __syncthreads();
    for (int s = 128; s > 0; s >>= 1) {
        if (tid < s) red[tid] += red[tid + s];
        __syncthreads();
    }
    const float inv = 1.f / red[0];
    #pragma unroll
    for (int i = 0; i < 32; ++i)
        row[tid + 256 * i] = vals[i] * inv;
}

extern "C" void kernel_launch(void* const* d_in, const int* in_sizes, int n_in,
                              void* d_out, int out_size, void* d_ws, size_t ws_size,
                              hipStream_t stream) {
    const float* enc    = (const float*)d_in[0];
    const float* h0     = (const float*)d_in[1];
    const float* c0     = (const float*)d_in[2];
    const float* st     = (const float*)d_in[3];
    const float* W_k    = (const float*)d_in[4];
    const float* W_r    = (const float*)d_in[5];
    const float* b_lstm = (const float*)d_in[6];
    const float* W_e    = (const float*)d_in[7];
    const float* b_e    = (const float*)d_in[8];
    const float* W_d    = (const float*)d_in[9];
    const float* b_d    = (const float*)d_in[10];
    const float* v_w    = (const float*)d_in[11];
    const float* v_b    = (const float*)d_in[12];
    float* out = (float*)d_out;

    float*    dec = (float*)((char*)d_ws + WS_DEC_OFF);      // 32KB
    _Float16* Wt  = (_Float16*)((char*)d_ws + WS_WT_OFF);    // 135KB

    wprep_kernel<<<256, 256, 0, stream>>>(W_e, Wt);
    lstm_dec_kernel<<<32, 256, 0, stream>>>(h0, c0, st, W_k, W_r, b_lstm, W_d, b_d, dec);
    attn_gemm_kernel<<<dim3(64, 32), 256, SMEM_BYTES, stream>>>(enc, Wt, b_e, v_w, v_b, dec, out);
    softmax_kernel<<<32, 256, 0, stream>>>(out);
}
